// Predictor_87050397155487
// MI455X (gfx1250) — matrix operand order repaired and checked by reading, not compiled, not run on hardware
//
#include <hip/hip_runtime.h>
#include <hip/hip_bf16.h>

#define HID   1024
#define DIN   256
#define BATCH 64
#define SEQ   96
#define HOR   96

typedef __attribute__((ext_vector_type(16))) __bf16       v16bf;
typedef __attribute__((ext_vector_type(8)))  float        v8f;
typedef __attribute__((ext_vector_type(4)))  unsigned int v4u;
typedef __attribute__((ext_vector_type(8)))  unsigned int v8u;

__device__ __forceinline__ unsigned short f32_to_bf16(float f) {
  unsigned int u = __builtin_bit_cast(unsigned int, f);
  u += 0x7FFFu + ((u >> 16) & 1u);          // round-to-nearest-even
  return (unsigned short)(u >> 16);
}

__device__ __forceinline__ float sigf(float x) {
  return 1.0f / (1.0f + __expf(-x));
}
__device__ __forceinline__ float tanhfast(float x) {
  return 1.0f - 2.0f / (__expf(2.0f * x) + 1.0f);
}

__device__ __forceinline__ v8f wmma_bf16(v16bf a, v8u b, v8f c) {
  return __builtin_amdgcn_wmma_f32_16x16x32_bf16(
      false, a, false, __builtin_bit_cast(v16bf, b), (short)0, c, false, false);
}

// Load 16-bit A fragment (16x32, M=lane%16): 16 contiguous halves at k (8 at k, 8 at k+8).
__device__ __forceinline__ v16bf load_a_frag(const unsigned short* ap) {
  v4u alo = *(const v4u*)ap;
  v4u ahi = *(const v4u*)(ap + 8);
  v8u au = {alo[0], alo[1], alo[2], alo[3], ahi[0], ahi[1], ahi[2], ahi[3]};
  return __builtin_bit_cast(v16bf, au);
}

// One K-segment of the 4-gate GEMM: marching pointers, compile-time trip count.
template <int KSEG>
__device__ __forceinline__ void gemm_quad(
    const unsigned short* __restrict__ ap,    // A row ptr (already + akb)
    const unsigned short* __restrict__ b0,    // W rows for gates i/f/g/o
    const unsigned short* __restrict__ b1,    //   (already + bkb)
    const unsigned short* __restrict__ b2,
    const unsigned short* __restrict__ b3,
    v8f& acc0, v8f& acc1, v8f& acc2, v8f& acc3)
{
#pragma unroll 4
  for (int k = 0; k < KSEG; k += 32) {
    v16bf a = load_a_frag(ap + k);
    if (k + 32 < KSEG) {
      __builtin_prefetch(b0 + k + 32, 0, 3);   // stage next weight chunk nearer
      __builtin_prefetch(b2 + k + 32, 0, 3);
    }
    v8u vb0 = *(const v8u*)(b0 + k);
    v8u vb1 = *(const v8u*)(b1 + k);
    v8u vb2 = *(const v8u*)(b2 + k);
    v8u vb3 = *(const v8u*)(b3 + k);
    acc0 = wmma_bf16(a, vb0, acc0);
    acc1 = wmma_bf16(a, vb1, acc1);
    acc2 = wmma_bf16(a, vb2, acc2);
    acc3 = wmma_bf16(a, vb3, acc3);
  }
}

// ---------------------------------------------------------------------------
// f32 -> bf16 conversion
__global__ void k_f32_to_bf16(const float* __restrict__ in,
                              unsigned short* __restrict__ out, int n) {
  int i = blockIdx.x * blockDim.x + threadIdx.x;
  if (i < n) out[i] = f32_to_bf16(in[i]);
}

__global__ void k_zero(unsigned int* __restrict__ p, int n) {
  int i = blockIdx.x * blockDim.x + threadIdx.x;
  if (i < n) p[i] = 0u;
}

// ---------------------------------------------------------------------------
// Fused LSTM cell step: gates = [A0|A1] @ [Wih|Whh]^T + (b_ih+b_hh), then
// c' = sig(f)*c + sig(i)*tanh(g); h' = sig(o)*tanh(c').
// Grid: HID/16 blocks (16-wide hidden strip). Block: 128 thr = 4 waves,
// wave w owns M-tile w (rows 16w..16w+15) and all 4 gate tiles for the strip.
template <int K0, int LDA0, int K1, int LDA1>
__global__ __launch_bounds__(128) void k_lstm_cell(
    const unsigned short* __restrict__ A0,   // x segment   (B x K0) bf16
    const unsigned short* __restrict__ A1,   // h_prev seg  (B x K1) bf16
    const unsigned short* __restrict__ Wih,  // (4H x K0) bf16
    const unsigned short* __restrict__ Whh,  // (4H x K1) bf16
    const float* __restrict__ b_ih, const float* __restrict__ b_hh,
    float* __restrict__ Cst,                 // (B x H) f32
    unsigned short* __restrict__ Hout)       // (B x H) bf16
{
  const int tid  = threadIdx.x;
  const int wave = tid >> 5;
  const int lane = tid & 31;
  const int col  = lane & 15;
  const int half = lane >> 4;
  const int j    = (blockIdx.x << 4) + col;   // hidden column
  const int am   = (wave << 4) + col;         // A-matrix row (batch index)
  const int akb  = half ? 16 : 0;
  const int bkb  = half ? 16 : 0;

  v8f z = {0.f, 0.f, 0.f, 0.f, 0.f, 0.f, 0.f, 0.f};
  v8f acc0 = z, acc1 = z, acc2 = z, acc3 = z;

  // Hoisted base pointers (constant strides; no per-iter 64-bit address math).
  const unsigned short* a0p = A0 + (size_t)am * LDA0 + akb;
  const unsigned short* a1p = A1 + (size_t)am * LDA1 + akb;
  const unsigned short* wi0 = Wih + (size_t)(0 * HID + j) * K0 + bkb;
  const unsigned short* wi1 = Wih + (size_t)(1 * HID + j) * K0 + bkb;
  const unsigned short* wi2 = Wih + (size_t)(2 * HID + j) * K0 + bkb;
  const unsigned short* wi3 = Wih + (size_t)(3 * HID + j) * K0 + bkb;
  const unsigned short* wh0 = Whh + (size_t)(0 * HID + j) * K1 + bkb;
  const unsigned short* wh1 = Whh + (size_t)(1 * HID + j) * K1 + bkb;
  const unsigned short* wh2 = Whh + (size_t)(2 * HID + j) * K1 + bkb;
  const unsigned short* wh3 = Whh + (size_t)(3 * HID + j) * K1 + bkb;

  gemm_quad<K0>(a0p, wi0, wi1, wi2, wi3, acc0, acc1, acc2, acc3);
  gemm_quad<K1>(a1p, wh0, wh1, wh2, wh3, acc0, acc1, acc2, acc3);

  const float bi = b_ih[0 * HID + j] + b_hh[0 * HID + j];
  const float bf = b_ih[1 * HID + j] + b_hh[1 * HID + j];
  const float bg = b_ih[2 * HID + j] + b_hh[2 * HID + j];
  const float bo = b_ih[3 * HID + j] + b_hh[3 * HID + j];

  const int mbase = (wave << 4) + (half << 3);
#pragma unroll
  for (int r = 0; r < 8; ++r) {
    const int m = mbase + r;
    const size_t idx = (size_t)m * HID + j;
    const float iv = acc0[r] + bi;
    const float fv = acc1[r] + bf;
    const float gv = acc2[r] + bg;
    const float ov = acc3[r] + bo;
    const float cp = Cst[idx];
    const float cn = sigf(fv) * cp + sigf(iv) * tanhfast(gv);
    const float hn = sigf(ov) * tanhfast(cn);
    Cst[idx]  = cn;
    Hout[idx] = f32_to_bf16(hn);
  }
}

// ---------------------------------------------------------------------------
// Decoder: logits = H @ Wdec^T + b_dec (64x256), softmax over 256, write f32
// probs to out and bf16 probs to pbuf (next AR input). One block, 16 waves.
// Softmax: 8 lanes per row, shfl_xor reductions within aligned 8-lane groups.
__global__ __launch_bounds__(512) void k_decode_softmax(
    const unsigned short* __restrict__ H,      // (B x H) bf16
    const unsigned short* __restrict__ Wdec,   // (DIN x H) bf16
    const float* __restrict__ bdec,
    float* __restrict__ outp,                  // (B x DIN) f32
    unsigned short* __restrict__ pbuf)         // (B x DIN) bf16
{
  __shared__ float logits[BATCH * DIN];        // 64 KB

  const int tid   = threadIdx.x;
  const int wave  = tid >> 5;
  const int lane  = tid & 31;
  const int col   = lane & 15;
  const int half  = lane >> 4;
  const int mtile = wave & 3;
  const int ntb   = (wave >> 2) << 2;          // 4 consecutive N-tiles per wave
  const int akb   = half ? 16 : 0;
  const int bkb   = half ? 16 : 0;

  const unsigned short* ap = H + (size_t)((mtile << 4) + col) * HID + akb;
  const unsigned short* b0 = Wdec + (size_t)(((ntb + 0) << 4) + col) * HID + bkb;
  const unsigned short* b1 = Wdec + (size_t)(((ntb + 1) << 4) + col) * HID + bkb;
  const unsigned short* b2 = Wdec + (size_t)(((ntb + 2) << 4) + col) * HID + bkb;
  const unsigned short* b3 = Wdec + (size_t)(((ntb + 3) << 4) + col) * HID + bkb;

  v8f z = {0.f, 0.f, 0.f, 0.f, 0.f, 0.f, 0.f, 0.f};
  v8f acc0 = z, acc1 = z, acc2 = z, acc3 = z;
  gemm_quad<HID>(ap, b0, b1, b2, b3, acc0, acc1, acc2, acc3);

  v8f acc[4] = {acc0, acc1, acc2, acc3};
#pragma unroll
  for (int q = 0; q < 4; ++q) {
    const int n = ((ntb + q) << 4) + col;
    const float bd = bdec[n];
#pragma unroll
    for (int r = 0; r < 8; ++r) {
      const int m = (mtile << 4) + (half << 3) + r;
      logits[m * DIN + n] = acc[q][r] + bd;
    }
  }
  __syncthreads();

  // Parallel softmax: 512 threads, 8 per row (aligned 8-lane groups).
  {
    const int row = tid >> 3;                 // 0..63
    const int sub = tid & 7;                  // 0..7
    const float* L = &logits[row * DIN];

    float mx = -1e30f;
#pragma unroll 8
    for (int c = sub; c < DIN; c += 8) mx = fmaxf(mx, L[c]);
    mx = fmaxf(mx, __shfl_xor(mx, 1));
    mx = fmaxf(mx, __shfl_xor(mx, 2));
    mx = fmaxf(mx, __shfl_xor(mx, 4));

    float s = 0.f;
#pragma unroll 8
    for (int c = sub; c < DIN; c += 8) s += __expf(L[c] - mx);
    s += __shfl_xor(s, 1);
    s += __shfl_xor(s, 2);
    s += __shfl_xor(s, 4);
    const float inv = 1.0f / s;

#pragma unroll 8
    for (int c = sub; c < DIN; c += 8) {
      const float p = __expf(L[c] - mx) * inv;
      outp[(size_t)row * DIN + c] = p;
      pbuf[row * DIN + c] = f32_to_bf16(p);
    }
  }
}

// ---------------------------------------------------------------------------
extern "C" void kernel_launch(void* const* d_in, const int* in_sizes, int n_in,
                              void* d_out, int out_size, void* d_ws, size_t ws_size,
                              hipStream_t stream) {
  (void)in_sizes; (void)n_in; (void)out_size; (void)ws_size;

  const float* X     = (const float*)d_in[0];
  const float* W_ih0 = (const float*)d_in[1];
  const float* W_hh0 = (const float*)d_in[2];
  const float* b_ih0 = (const float*)d_in[3];
  const float* b_hh0 = (const float*)d_in[4];
  const float* W_ih1 = (const float*)d_in[5];
  const float* W_hh1 = (const float*)d_in[6];
  const float* b_ih1 = (const float*)d_in[7];
  const float* b_hh1 = (const float*)d_in[8];
  const float* W_dec = (const float*)d_in[9];
  const float* b_dec = (const float*)d_in[10];

  char* ws = (char*)d_ws;
  auto take = [&](size_t bytes) -> char* {
    char* p = ws;
    ws += (bytes + 255) & ~(size_t)255;
    return p;
  };

  unsigned short* Xb    = (unsigned short*)take((size_t)SEQ * BATCH * DIN * 2);
  unsigned short* Wih0b = (unsigned short*)take((size_t)4 * HID * DIN * 2);
  unsigned short* Whh0b = (unsigned short*)take((size_t)4 * HID * HID * 2);
  unsigned short* Wih1b = (unsigned short*)take((size_t)4 * HID * HID * 2);
  unsigned short* Whh1b = (unsigned short*)take((size_t)4 * HID * HID * 2);
  unsigned short* Wdecb = (unsigned short*)take((size_t)DIN * HID * 2);
  unsigned short* h0a   = (unsigned short*)take((size_t)BATCH * HID * 2);
  unsigned short* h0b   = (unsigned short*)take((size_t)BATCH * HID * 2);
  unsigned short* h1a   = (unsigned short*)take((size_t)BATCH * HID * 2);
  unsigned short* h1b   = (unsigned short*)take((size_t)BATCH * HID * 2);
  float*          c0    = (float*)take((size_t)BATCH * HID * 4);
  float*          c1    = (float*)take((size_t)BATCH * HID * 4);
  unsigned short* pbuf  = (unsigned short*)take((size_t)BATCH * DIN * 2);

  auto cvt = [&](const float* src, unsigned short* dst, int n) {
    k_f32_to_bf16<<<(n + 255) / 256, 256, 0, stream>>>(src, dst, n);
  };
  cvt(X,     Xb,    SEQ * BATCH * DIN);
  cvt(W_ih0, Wih0b, 4 * HID * DIN);
  cvt(W_hh0, Whh0b, 4 * HID * HID);
  cvt(W_ih1, Wih1b, 4 * HID * HID);
  cvt(W_hh1, Whh1b, 4 * HID * HID);
  cvt(W_dec, Wdecb, DIN * HID);

  // zero initial hidden / cell state (parity-0 h buffers are read at t=0)
  const int hw = BATCH * HID / 2;   // u32 words in a bf16 h buffer
  const int cw = BATCH * HID;       // u32 words in an f32 c buffer
  k_zero<<<(hw + 255) / 256, 256, 0, stream>>>((unsigned int*)h0a, hw);
  k_zero<<<(hw + 255) / 256, 256, 0, stream>>>((unsigned int*)h1a, hw);
  k_zero<<<(cw + 255) / 256, 256, 0, stream>>>((unsigned int*)c0, cw);
  k_zero<<<(cw + 255) / 256, 256, 0, stream>>>((unsigned int*)c1, cw);

  unsigned short* h0[2] = {h0a, h0b};
  unsigned short* h1[2] = {h1a, h1b};
  float* out = (float*)d_out;

  for (int t = 0; t < SEQ + HOR; ++t) {
    const int rd = t & 1, wr = rd ^ 1;
    const unsigned short* x = (t < SEQ) ? (Xb + (size_t)t * BATCH * DIN) : pbuf;

    // layer 0: x(256) | h0(1024) -> h0'
    k_lstm_cell<DIN, DIN, HID, HID><<<HID / 16, 128, 0, stream>>>(
        x, h0[rd], Wih0b, Whh0b, b_ih0, b_hh0, c0, h0[wr]);

    // layer 1: h0'(1024) | h1(1024) -> h1'
    k_lstm_cell<HID, HID, HID, HID><<<HID / 16, 128, 0, stream>>>(
        h0[wr], h1[rd], Wih1b, Whh1b, b_ih1, b_hh1, c1, h1[wr]);

    // decode + softmax; also emits bf16 prediction for the next AR input
    k_decode_softmax<<<1, 512, 0, stream>>>(
        h1[wr], Wdecb, b_dec, out + (size_t)t * BATCH * DIN, pbuf);
  }
}
